// DeepLog_76579266888428
// MI455X (gfx1250) — compile-verified, matrix operand order checked by reading, not run on hardware
//
#include <hip/hip_runtime.h>
#include <hip/hip_bf16.h>
#include <stdint.h>

// ---------------------------------------------------------------------------
// Types
// ---------------------------------------------------------------------------
typedef __bf16 bf16_t;
typedef __attribute__((ext_vector_type(16))) __bf16 v16bf;
typedef __attribute__((ext_vector_type(8)))  __bf16 v8bf;
typedef __attribute__((ext_vector_type(8)))  float  v8f;
typedef __attribute__((ext_vector_type(4)))  unsigned int u32x4;
typedef __attribute__((ext_vector_type(8)))  int    i32x8;
typedef __attribute__((ext_vector_type(4)))  int    i32x4;

#define BATCH   128
#define SEQ     256
#define VOCAB   1024
#define HID     512
#define GATES   2048      // 4*HID
#define OUTN    1024
#define KCHUNK  256       // K staged per LDS phase
#define LDSPAD  16        // bf16 elems of pad per LDS row (32B = 8 dwords)
#define BPITCH  (KCHUNK + LDSPAD)

// ---------------------------------------------------------------------------
// WMMA fragment loaders (16x16x32 bf16; A/B: 8 VGPRs per lane)
// ISA layout: lanes 0-15 hold K = kb+0..7 and kb+16..23; lanes 16-31 hold
// K = kb+8..15 and kb+24..31 (rows/cols = lane & 15).
// ---------------------------------------------------------------------------
union FragU { v16bf v; v8bf h[2]; };

__device__ __forceinline__ v16bf load_frag(const bf16_t* __restrict__ base,
                                           int pitch, int row, int kbase, int lane) {
    const int kb = kbase + ((lane >> 4) << 3);
    const bf16_t* p = base + (size_t)row * pitch + kb;
    FragU u;
    u.h[0] = *(const v8bf*)(p);
    u.h[1] = *(const v8bf*)(p + 16);
    return u.v;
}

__device__ __forceinline__ float sigmoidf_(float x) {
    return 1.0f / (1.0f + __expf(-x));
}

// ---------------------------------------------------------------------------
// Tensor Data Mover: 2D tile [16 rows x KCHUNK bf16] from a [rows x HID] bf16
// tensor into LDS, with LDS padding (interval 512B -> code 6, amount 8 dwords
// -> code 7).  D# packing per CDNA5 ISA §8.3/8.4.
// This toolchain exposes the 6-arg builtin:
//   (u32x4 g0, i32x8 g1, i32x4 g2, i32x4 g3, i32x8 g45, i32 cpol)
// Groups 2..n are zero: 2D tensor (tile_dim2..4 = 0 -> unused).
// ---------------------------------------------------------------------------
__device__ __forceinline__ void tdm_load_tile(unsigned lds_off,
                                              const bf16_t* gsrc,
                                              unsigned tensor_rows) {
    unsigned long long ga = (unsigned long long)(uintptr_t)gsrc;
    u32x4 g0 = {
        0x1u,                                    // count=1, user mode
        lds_off,                                 // lds_addr (bytes)
        (unsigned)(ga & 0xFFFFFFFFu),            // global_addr[31:0]
        (unsigned)((ga >> 32) & 0x1FFFFFFu) | (2u << 30)  // addr[56:32] | type=2
    };
    const unsigned w0 = (1u << 16)               // data_size = 1 (2 bytes)
                      | (1u << 20)               // pad_enable
                      | (6u << 22)               // pad_interval: 128 dwords = 512B
                      | (7u << 25);              // pad_amount: 8 dwords = 32B
    i32x8 g1 = {
        (int)w0,
        (int)((HID & 0xFFFFu) << 16),            // tensor_dim0[15:0]=512 (atomic addr=0)
        (int)(((HID >> 16) & 0xFFFFu) | ((tensor_rows & 0xFFFFu) << 16)),
        (int)(((tensor_rows >> 16) & 0xFFFFu) | ((unsigned)KCHUNK << 16)), // tile_dim0
        (int)(16u),                              // tile_dim1=16, tile_dim2=0
        (int)(HID),                              // tensor_dim0_stride[31:0]=512
        0,                                       // stride0 hi | stride1 lo
        0
    };
    i32x4 z4 = {0, 0, 0, 0};
    i32x8 z8 = {0, 0, 0, 0, 0, 0, 0, 0};
    __builtin_amdgcn_tensor_load_to_lds(g0, g1, z4, z4, z8, 0);
}

// ---------------------------------------------------------------------------
// One-time: transpose + fp32 -> bf16  (out[n*K + k] = in[k*N + n])
// ---------------------------------------------------------------------------
__global__ void convert_transpose(const float* __restrict__ in,
                                  bf16_t* __restrict__ out, int K, int N) {
    int idx = blockIdx.x * blockDim.x + threadIdx.x;
    if (idx >= K * N) return;
    int n = idx / K, k = idx - n * K;
    out[idx] = (bf16_t)in[k * N + n];
}

__global__ void zero_words(unsigned int* __restrict__ p, int n) {
    int idx = blockIdx.x * blockDim.x + threadIdx.x;
    if (idx < n) p[idx] = 0u;
}

// ---------------------------------------------------------------------------
// LSTM layer 1, one timestep.  z = gather(Wx1, x_t) + h1 @ Wh1 + b1 ; gates.
// Grid: 32 blocks (gate-column tiles), 256 threads = 8 waves (M tiles).
// ---------------------------------------------------------------------------
__global__ void __launch_bounds__(256)
lstm_layer1_step(const int* __restrict__ x, int t,
                 const float* __restrict__ Wx1, const float* __restrict__ b1,
                 const bf16_t* __restrict__ Wh1T,
                 const bf16_t* __restrict__ h_in, bf16_t* __restrict__ h_out,
                 float* __restrict__ c) {
    __shared__ bf16_t smem[4][16][BPITCH];
    const int lane = threadIdx.x & 31;
    const int wave = threadIdx.x >> 5;
    const int n0   = blockIdx.x * 16;   // column tile within one gate (H=512)
    const int m0   = wave * 16;         // 8 waves cover M = 128

    v8f acc[4] = {};
    for (int kc = 0; kc < HID; kc += KCHUNK) {
        __syncthreads();
        if (wave == 0) {
            #pragma unroll
            for (int g = 0; g < 4; ++g) {
                unsigned lds_off = (unsigned)(uintptr_t)(&smem[g][0][0]);
                tdm_load_tile(lds_off, Wh1T + (size_t)(g * HID + n0) * HID + kc, GATES);
            }
        }
        __builtin_amdgcn_s_wait_tensorcnt(0);
        __syncthreads();
        #pragma unroll
        for (int ks = 0; ks < KCHUNK; ks += 32) {
            v16bf a = load_frag(h_in, HID, m0 + (lane & 15), kc + ks, lane);
            #pragma unroll
            for (int g = 0; g < 4; ++g) {
                v16bf b = load_frag(&smem[g][0][0], BPITCH, lane & 15, ks, lane);
                acc[g] = __builtin_amdgcn_wmma_f32_16x16x32_bf16(
                    false, a, false, b, (short)0, acc[g], false, false);
            }
        }
    }
    // Fused epilogue: one-hot gather + bias + gates + cell update + bf16 h.
    const int n     = lane & 15;
    const int hcol  = n0 + n;
    const int mbase = m0 + ((lane >> 4) << 3);
    #pragma unroll
    for (int r = 0; r < 8; ++r) {
        const int m = mbase + r;
        const int xi = x[m * SEQ + t];
        const float* xrow = Wx1 + (size_t)xi * GATES;
        float zi = acc[0][r] + b1[hcol]            + xrow[hcol];
        float zf = acc[1][r] + b1[HID + hcol]      + xrow[HID + hcol];
        float zg = acc[2][r] + b1[2 * HID + hcol]  + xrow[2 * HID + hcol];
        float zo = acc[3][r] + b1[3 * HID + hcol]  + xrow[3 * HID + hcol];
        float ii = sigmoidf_(zi), ff = sigmoidf_(zf);
        float gg = tanhf(zg),     oo = sigmoidf_(zo);
        float cn = ff * c[m * HID + hcol] + ii * gg;
        c[m * HID + hcol]     = cn;
        h_out[m * HID + hcol] = (bf16_t)(oo * tanhf(cn));
    }
}

// ---------------------------------------------------------------------------
// LSTM layer 2, one timestep.  z = h1 @ Wx2 + h2 @ Wh2 + b2 ; gates.
// ---------------------------------------------------------------------------
__global__ void __launch_bounds__(256)
lstm_layer2_step(const float* __restrict__ b2,
                 const bf16_t* __restrict__ Wx2T, const bf16_t* __restrict__ Wh2T,
                 const bf16_t* __restrict__ h1_in,
                 const bf16_t* __restrict__ h2_in, bf16_t* __restrict__ h2_out,
                 float* __restrict__ c) {
    __shared__ bf16_t smem[4][16][BPITCH];
    const int lane = threadIdx.x & 31;
    const int wave = threadIdx.x >> 5;
    const int n0   = blockIdx.x * 16;
    const int m0   = wave * 16;

    v8f acc[4] = {};
    for (int src = 0; src < 2; ++src) {
        const bf16_t* Ain = src ? h2_in : h1_in;
        const bf16_t* Wt  = src ? Wh2T  : Wx2T;
        for (int kc = 0; kc < HID; kc += KCHUNK) {
            __syncthreads();
            if (wave == 0) {
                #pragma unroll
                for (int g = 0; g < 4; ++g) {
                    unsigned lds_off = (unsigned)(uintptr_t)(&smem[g][0][0]);
                    tdm_load_tile(lds_off, Wt + (size_t)(g * HID + n0) * HID + kc, GATES);
                }
            }
            __builtin_amdgcn_s_wait_tensorcnt(0);
            __syncthreads();
            #pragma unroll
            for (int ks = 0; ks < KCHUNK; ks += 32) {
                v16bf a = load_frag(Ain, HID, m0 + (lane & 15), kc + ks, lane);
                #pragma unroll
                for (int g = 0; g < 4; ++g) {
                    v16bf b = load_frag(&smem[g][0][0], BPITCH, lane & 15, ks, lane);
                    acc[g] = __builtin_amdgcn_wmma_f32_16x16x32_bf16(
                        false, a, false, b, (short)0, acc[g], false, false);
                }
            }
        }
    }
    const int n     = lane & 15;
    const int hcol  = n0 + n;
    const int mbase = m0 + ((lane >> 4) << 3);
    #pragma unroll
    for (int r = 0; r < 8; ++r) {
        const int m = mbase + r;
        float zi = acc[0][r] + b2[hcol];
        float zf = acc[1][r] + b2[HID + hcol];
        float zg = acc[2][r] + b2[2 * HID + hcol];
        float zo = acc[3][r] + b2[3 * HID + hcol];
        float ii = sigmoidf_(zi), ff = sigmoidf_(zf);
        float gg = tanhf(zg),     oo = sigmoidf_(zo);
        float cn = ff * c[m * HID + hcol] + ii * gg;
        c[m * HID + hcol]      = cn;
        h2_out[m * HID + hcol] = (bf16_t)(oo * tanhf(cn));
    }
}

// ---------------------------------------------------------------------------
// Final dense: logits = h2 @ Wd + bd   (128 x 512 x 1024), WMMA bf16.
// Grid: 64 blocks x 8 waves; one 16x16 tile per wave; B direct from L2.
// ---------------------------------------------------------------------------
__global__ void __launch_bounds__(256)
dense_logits(const bf16_t* __restrict__ h2, const bf16_t* __restrict__ WdT,
             const float* __restrict__ bd, float* __restrict__ logits) {
    const int lane = threadIdx.x & 31;
    const int wave = threadIdx.x >> 5;
    const int n0   = blockIdx.x * 16;
    const int m0   = wave * 16;
    v8f acc = {};
    #pragma unroll
    for (int kc = 0; kc < HID; kc += 32) {
        v16bf a = load_frag(h2,  HID, m0 + (lane & 15), kc, lane);
        v16bf b = load_frag(WdT, HID, n0 + (lane & 15), kc, lane);
        acc = __builtin_amdgcn_wmma_f32_16x16x32_bf16(
            false, a, false, b, (short)0, acc, false, false);
    }
    const int n     = n0 + (lane & 15);
    const int mbase = m0 + ((lane >> 4) << 3);
    #pragma unroll
    for (int r = 0; r < 8; ++r)
        logits[(mbase + r) * OUTN + n] = acc[r] + bd[n];
}

// ---------------------------------------------------------------------------
// Row softmax over [128, 1024].
// ---------------------------------------------------------------------------
__global__ void __launch_bounds__(256)
softmax_rows(const float* __restrict__ logits, float* __restrict__ out) {
    __shared__ float red[256];
    const int row = blockIdx.x, tid = threadIdx.x;
    const float* lr = logits + (size_t)row * OUTN;
    float m = -1e30f;
    for (int j = tid; j < OUTN; j += 256) m = fmaxf(m, lr[j]);
    red[tid] = m; __syncthreads();
    for (int s = 128; s > 0; s >>= 1) {
        if (tid < s) red[tid] = fmaxf(red[tid], red[tid + s]);
        __syncthreads();
    }
    const float mx = red[0]; __syncthreads();
    float sum = 0.0f;
    for (int j = tid; j < OUTN; j += 256) sum += __expf(lr[j] - mx);
    red[tid] = sum; __syncthreads();
    for (int s = 128; s > 0; s >>= 1) {
        if (tid < s) red[tid] += red[tid + s];
        __syncthreads();
    }
    const float inv = 1.0f / red[0];
    for (int j = tid; j < OUTN; j += 256)
        out[(size_t)row * OUTN + j] = __expf(lr[j] - mx) * inv;
}

// ---------------------------------------------------------------------------
// Launch
// ---------------------------------------------------------------------------
extern "C" void kernel_launch(void* const* d_in, const int* in_sizes, int n_in,
                              void* d_out, int out_size, void* d_ws, size_t ws_size,
                              hipStream_t stream) {
    const int*   x   = (const int*)  d_in[0];
    const float* Wx1 = (const float*)d_in[1];
    const float* Wh1 = (const float*)d_in[2];
    const float* b1  = (const float*)d_in[3];
    const float* Wx2 = (const float*)d_in[4];
    const float* Wh2 = (const float*)d_in[5];
    const float* b2  = (const float*)d_in[6];
    const float* Wd  = (const float*)d_in[7];
    const float* bd  = (const float*)d_in[8];
    float* out = (float*)d_out;

    // Workspace layout
    char* p = (char*)d_ws;
    bf16_t* Wh1T = (bf16_t*)p; p += (size_t)GATES * HID * sizeof(bf16_t);
    bf16_t* Wx2T = (bf16_t*)p; p += (size_t)GATES * HID * sizeof(bf16_t);
    bf16_t* Wh2T = (bf16_t*)p; p += (size_t)GATES * HID * sizeof(bf16_t);
    bf16_t* WdT  = (bf16_t*)p; p += (size_t)OUTN  * HID * sizeof(bf16_t);
    char* state  = p;
    float*  c1   = (float*)p;  p += (size_t)BATCH * HID * sizeof(float);
    float*  c2   = (float*)p;  p += (size_t)BATCH * HID * sizeof(float);
    bf16_t* h1b[2]; bf16_t* h2b[2];
    h1b[0] = (bf16_t*)p; p += (size_t)BATCH * HID * sizeof(bf16_t);
    h1b[1] = (bf16_t*)p; p += (size_t)BATCH * HID * sizeof(bf16_t);
    h2b[0] = (bf16_t*)p; p += (size_t)BATCH * HID * sizeof(bf16_t);
    h2b[1] = (bf16_t*)p; p += (size_t)BATCH * HID * sizeof(bf16_t);
    const int state_words = (int)((p - state) / 4);
    float* logits = (float*)p; p += (size_t)BATCH * OUTN * sizeof(float);

    // 0) zero recurrent state (every call: deterministic)
    zero_words<<<(state_words + 255) / 256, 256, 0, stream>>>(
        (unsigned int*)state, state_words);

    // 1) weight transpose + bf16 convert (L2-resident afterwards)
    {
        int tot = GATES * HID;
        convert_transpose<<<(tot + 255) / 256, 256, 0, stream>>>(Wh1, Wh1T, HID, GATES);
        convert_transpose<<<(tot + 255) / 256, 256, 0, stream>>>(Wx2, Wx2T, HID, GATES);
        convert_transpose<<<(tot + 255) / 256, 256, 0, stream>>>(Wh2, Wh2T, HID, GATES);
        int totd = OUTN * HID;
        convert_transpose<<<(totd + 255) / 256, 256, 0, stream>>>(Wd, WdT, HID, OUTN);
    }

    // 2) sequential recurrence: 2 fused WMMA kernels per timestep
    for (int t = 0; t < SEQ; ++t) {
        const int cur = t & 1, prv = cur ^ 1;
        lstm_layer1_step<<<HID / 16, 256, 0, stream>>>(
            x, t, Wx1, b1, Wh1T, h1b[prv], h1b[cur], c1);
        lstm_layer2_step<<<HID / 16, 256, 0, stream>>>(
            b2, Wx2T, Wh2T, h1b[cur], h2b[prv], h2b[cur], c2);
    }

    // 3) final dense + softmax
    dense_logits<<<OUTN / 16, 256, 0, stream>>>(h2b[(SEQ - 1) & 1], WdT, bd, logits);
    softmax_rows<<<BATCH, 256, 0, stream>>>(logits, out);
}
